// Bimamba_outer_21930103014088
// MI455X (gfx1250) — compile-verified, matrix operand order checked by reading, not run on hardware
//
#include <hip/hip_runtime.h>

// ============================================================================
// Bidirectional Mamba block for MI455X (gfx1250), wave32 + WMMA.
//
// All large GEMMs run through v_wmma_f32_16x16x32_bf16 (bf16 in, f32 acc).
// GEMM contract (ALL call sites satisfy it by construction/padding):
//   M % 128 == 0, N % 128 == 0, K % 32 == 0
// -> every fragment load, WMMA and store is unconditional: the inner loop is
//    branch-free (12x global_load_b128 + 2x global_prefetch_b8 + 8x v_wmma).
// The N=80 x_dbl projection is padded to N=128 (W_x rows 80..127 zeroed,
// xdbl stride 128; the pad columns are never read downstream).
//
// The sequential selective scan is a per-channel VALU kernel (16-state in
// registers, B/C vectors staged in LDS, exp via v_exp_f32).
//
// Input order assumption (jax pytree / alphabetical within each param dict):
//  0: hidden_input (2,2048,768)
//  params_f: 1:A_log 2:D 3:W_dt 4:W_in 5:W_out 6:W_x 7:b_dt 8:conv_b 9:conv_w
//  params_b: 10..18 same order
//  19: W_proj (768,1536)   20: b_proj (768)
// Workspace use: ~300 MB of d_ws.
// ============================================================================

#define BB 2
#define LL 2048
#define DM 768
#define DI 1536          // d_inner
#define DS 16            // d_state
#define RK 48            // dt_rank
#define MM (BB * LL)     // 4096 token rows
#define XDW 128          // padded x_dbl width (>= RK + 2*DS = 80, mult of 64)

typedef __attribute__((ext_vector_type(16))) __bf16 v16bf;
typedef __attribute__((ext_vector_type(8)))  __bf16 v8bf;
typedef __attribute__((ext_vector_type(8)))  float  v8f;

// ---------------------------------------------------------------------------
// fp32 -> bf16 conversion
// ---------------------------------------------------------------------------
__global__ void f32_to_bf16_k(const float* __restrict__ src,
                              __bf16* __restrict__ dst, long long n) {
  long long i = (long long)blockIdx.x * 256 + threadIdx.x;
  if (i < n) dst[i] = (__bf16)src[i];
}

// convert with row+column zero-padding:
//   dst[r, c] (r < rows, c < dcols) = (r < srows && c < scols) ? src[r*src_ld+c] : 0
// used for: W_dt (48 -> 64 cols), W_x (80 -> 128 rows), dt_r slice of xdbl.
__global__ void pad_cvt_k(const float* __restrict__ src, __bf16* __restrict__ dst,
                          int rows, int src_ld, int srows, int scols, int dcols) {
  long long i = (long long)blockIdx.x * 256 + threadIdx.x;
  long long total = (long long)rows * dcols;
  if (i >= total) return;
  int r = (int)(i / dcols), c = (int)(i % dcols);
  float v = (r < srows && c < scols) ? src[(long long)r * src_ld + c] : 0.0f;
  dst[i] = (__bf16)v;
}

// ---------------------------------------------------------------------------
// WMMA bf16 GEMM:  C[M,N] = A[M,K] * Bw[N,K]^T (+bias), C fp32 or bf16.
// Workgroup: 256 threads = 8 waves as 4(M) x 2(N); wave tile 32x64 via
// 2x4 accumulators of v_wmma_f32_16x16x32_bf16. Workgroup tile 128x128.
//
// Fragment layout (16-bit A/B, 16x32, wave32):
//   row (M or N) = lane&15, half h = lane>>4
//   lane holds K in [8h, 8h+8) and [16+8h, 16+8h+8)  -> two 16B loads.
// C/D layout: col = lane&15, VGPR r holds row r + 8*(lane>>4).
// ---------------------------------------------------------------------------
__device__ __forceinline__ v16bf load_frag(const __bf16* __restrict__ base,
                                           int row, int ld, int k0, int lane) {
  int r = row + (lane & 15);
  int h = lane >> 4;
  const __bf16* p = base + (long long)r * ld + k0;
  v8bf lo = *(const v8bf*)(p + h * 8);
  v8bf hi = *(const v8bf*)(p + 16 + h * 8);
  v16bf f;
#pragma unroll
  for (int i = 0; i < 8; ++i) { f[i] = lo[i]; f[i + 8] = hi[i]; }
  return f;
}

__global__ void gemm_bf16_wmma_k(const __bf16* __restrict__ A,
                                 const __bf16* __restrict__ Bw,
                                 void* __restrict__ Cout,
                                 const float* __restrict__ bias,
                                 int M, int N, int K,
                                 int lda, int ldb, int ldc, int out_bf16) {
  (void)M; (void)N;  // tiles are always fully valid (see contract above)
  const int lane = threadIdx.x & 31;
  const int wave = threadIdx.x >> 5;          // 0..7
  const int wm = wave & 3;                    // 0..3
  const int wn = wave >> 2;                   // 0..1
  const int tileM = blockIdx.y * 128 + wm * 32;
  const int tileN = blockIdx.x * 128 + wn * 64;

  v8f acc[2][4];
#pragma unroll
  for (int i = 0; i < 2; ++i)
#pragma unroll
    for (int j = 0; j < 4; ++j)
#pragma unroll
      for (int r = 0; r < 8; ++r) acc[i][j][r] = 0.0f;

#pragma unroll 2
  for (int k0 = 0; k0 < K; k0 += 32) {
    if (k0 + 128 < K) {  // gfx1250 global_prefetch_b8, 4 K-slabs ahead
      __builtin_prefetch(A + (long long)(tileM + (lane & 15)) * lda + k0 + 128, 0, 3);
      __builtin_prefetch(Bw + (long long)(tileN + (lane & 15)) * ldb + k0 + 128, 0, 3);
    }
    v16bf a0 = load_frag(A, tileM,      lda, k0, lane);
    v16bf a1 = load_frag(A, tileM + 16, lda, k0, lane);
    v16bf bfr[4];
#pragma unroll
    for (int j = 0; j < 4; ++j)
      bfr[j] = load_frag(Bw, tileN + j * 16, ldb, k0, lane);
#pragma unroll
    for (int j = 0; j < 4; ++j) {
      acc[0][j] = __builtin_amdgcn_wmma_f32_16x16x32_bf16(false, a0, false, bfr[j],
                                                          (short)0, acc[0][j], false, false);
      acc[1][j] = __builtin_amdgcn_wmma_f32_16x16x32_bf16(false, a1, false, bfr[j],
                                                          (short)0, acc[1][j], false, false);
    }
  }

  const int col = lane & 15;
  const int h = lane >> 4;
#pragma unroll
  for (int j = 0; j < 4; ++j) {
    int cg = tileN + j * 16 + col;
    float badd = bias ? bias[cg] : 0.0f;
#pragma unroll
    for (int i = 0; i < 2; ++i) {
#pragma unroll
      for (int r = 0; r < 8; ++r) {
        int rg = tileM + i * 16 + r + h * 8;
        float v = acc[i][j][r] + badd;
        if (out_bf16) ((__bf16*)Cout)[(long long)rg * ldc + cg] = (__bf16)v;
        else          ((float*)Cout)[(long long)rg * ldc + cg] = v;
      }
    }
  }
}

// ---------------------------------------------------------------------------
// Causal depthwise conv1d (taps=4) + SiLU, both directions in one launch.
// Scan-order output: xc_d[b,t,c] with source index idx(u)= (d? L-1-u : u).
// Writes fp32 (for scan) and bf16 (for the W_x GEMM).
// ---------------------------------------------------------------------------
__global__ void conv_silu_k(const float* __restrict__ xz_f, const float* __restrict__ xz_b,
                            const float* __restrict__ cw_f, const float* __restrict__ cb_f,
                            const float* __restrict__ cw_b, const float* __restrict__ cb_b,
                            float* __restrict__ xcf_f, float* __restrict__ xcf_b,
                            __bf16* __restrict__ xcb_f, __bf16* __restrict__ xcb_b) {
  long long i = (long long)blockIdx.x * 256 + threadIdx.x;
  const long long per = (long long)BB * LL * DI;
  if (i >= 2 * per) return;
  int d = (i >= per);
  long long j = i - (long long)d * per;
  int c = (int)(j % DI);
  long long bt = j / DI;
  int t = (int)(bt % LL);
  int b = (int)(bt / LL);
  const float* xz = d ? xz_b : xz_f;
  const float* cw = d ? cw_b : cw_f;
  const float* cb = d ? cb_b : cb_f;
  float s = cb[c];
#pragma unroll
  for (int jj = 0; jj < 4; ++jj) {
    int u = t - 3 + jj;
    if (u >= 0) {
      int l = d ? (LL - 1 - u) : u;
      s += cw[c * 4 + jj] * xz[((long long)b * LL + l) * (2 * DI) + c];
    }
  }
  float v = s / (1.0f + __expf(-s));   // silu
  (d ? xcf_b : xcf_f)[j] = v;
  (d ? xcb_b : xcb_f)[j] = (__bf16)v;
}

// ---------------------------------------------------------------------------
// Selective scan. grid = (DI/256, B, 2 dirs); one thread per channel.
// 16-wide state in registers; per-step B/C (uniform across channels) in LDS.
// ---------------------------------------------------------------------------
__global__ void scan_k(const float* __restrict__ xdbl_f, const float* __restrict__ xdbl_b,
                       const float* __restrict__ dtl_f,  const float* __restrict__ dtl_b,
                       const float* __restrict__ xcf_f,  const float* __restrict__ xcf_b,
                       const float* __restrict__ Alog_f, const float* __restrict__ bdt_f,
                       const float* __restrict__ Alog_b, const float* __restrict__ bdt_b,
                       float* __restrict__ ys_f, float* __restrict__ ys_b) {
  const int d = blockIdx.z;
  const int b = blockIdx.y;
  const int c = blockIdx.x * 256 + threadIdx.x;   // < DI by construction
  const float* xdbl = d ? xdbl_b : xdbl_f;
  const float* dtl  = d ? dtl_b  : dtl_f;
  const float* xcf  = d ? xcf_b  : xcf_f;
  const float* Alog = d ? Alog_b : Alog_f;
  const float* bdtp = d ? bdt_b  : bdt_f;
  float*       ys   = d ? ys_b   : ys_f;

  __shared__ float sBC[2 * DS];   // B then C
  float A[DS], h[DS];
#pragma unroll
  for (int n = 0; n < DS; ++n) { A[n] = -__expf(Alog[c * DS + n]); h[n] = 0.0f; }
  const float bdt = bdtp[c];

  for (int t = 0; t < LL; ++t) {
    long long bt = (long long)b * LL + t;
    if (threadIdx.x < 2 * DS)
      sBC[threadIdx.x] = xdbl[bt * XDW + RK + threadIdx.x];
    __syncthreads();
    float xv = dtl[bt * DI + c] + bdt;
    float dt = (xv > 20.0f) ? xv : __logf(1.0f + __expf(xv));   // softplus
    float dtu = dt * xcf[bt * DI + c];
    float y = 0.0f;
#pragma unroll
    for (int n = 0; n < DS; ++n) {
      h[n] = __expf(dt * A[n]) * h[n] + dtu * sBC[n];
      y += h[n] * sBC[DS + n];
    }
    ys[bt * DI + c] = y;
    __syncthreads();
  }
}

// ---------------------------------------------------------------------------
// y = (ys + xc*D) * silu(z), un-reversing the backward direction; bf16 out.
// ---------------------------------------------------------------------------
__global__ void gate_k(const float* __restrict__ ys_f, const float* __restrict__ ys_b,
                       const float* __restrict__ xcf_f, const float* __restrict__ xcf_b,
                       const float* __restrict__ xz_f, const float* __restrict__ xz_b,
                       const float* __restrict__ D_f, const float* __restrict__ D_b,
                       __bf16* __restrict__ ybf_f, __bf16* __restrict__ ybf_b) {
  long long i = (long long)blockIdx.x * 256 + threadIdx.x;
  const long long per = (long long)BB * LL * DI;
  if (i >= 2 * per) return;
  int d = (i >= per);
  long long j = i - (long long)d * per;
  int c = (int)(j % DI);
  long long bl = j / DI;
  int l = (int)(bl % LL);
  int b = (int)(bl / LL);
  int t = d ? (LL - 1 - l) : l;                       // scan-order index
  long long sc = ((long long)b * LL + t) * DI + c;
  const float* ys  = d ? ys_b  : ys_f;
  const float* xcf = d ? xcf_b : xcf_f;
  const float* xz  = d ? xz_b  : xz_f;
  const float* Dp  = d ? D_b   : D_f;
  float z = xz[((long long)b * LL + l) * (2 * DI) + DI + c];
  float y = (ys[sc] + xcf[sc] * Dp[c]) * (z / (1.0f + __expf(-z)));
  (d ? ybf_b : ybf_f)[j] = (__bf16)y;
}

// ===========================================================================
extern "C" void kernel_launch(void* const* d_in, const int* in_sizes, int n_in,
                              void* d_out, int out_size, void* d_ws, size_t ws_size,
                              hipStream_t stream) {
  (void)in_sizes; (void)n_in; (void)out_size; (void)ws_size;

  const float* x      = (const float*)d_in[0];
  const float* Alog_f = (const float*)d_in[1];
  const float* D_f    = (const float*)d_in[2];
  const float* Wdt_f  = (const float*)d_in[3];
  const float* Win_f  = (const float*)d_in[4];
  const float* Wout_f = (const float*)d_in[5];
  const float* Wx_f   = (const float*)d_in[6];
  const float* bdt_f  = (const float*)d_in[7];
  const float* cb_f   = (const float*)d_in[8];
  const float* cw_f   = (const float*)d_in[9];
  const float* Alog_b = (const float*)d_in[10];
  const float* D_b    = (const float*)d_in[11];
  const float* Wdt_b  = (const float*)d_in[12];
  const float* Win_b  = (const float*)d_in[13];
  const float* Wout_b = (const float*)d_in[14];
  const float* Wx_b   = (const float*)d_in[15];
  const float* bdt_b  = (const float*)d_in[16];
  const float* cb_b   = (const float*)d_in[17];
  const float* cw_b   = (const float*)d_in[18];
  const float* Wproj  = (const float*)d_in[19];
  const float* bproj  = (const float*)d_in[20];

  // --- workspace carve-up (256B aligned) ---
  size_t off = 0;
  auto alloc = [&](size_t nbytes) -> void* {
    void* p = (char*)d_ws + off;
    off = (off + nbytes + 255) & ~(size_t)255;
    return p;
  };
  __bf16* xbf      = (__bf16*)alloc((size_t)MM * DM * 2);
  __bf16* Winbf_f  = (__bf16*)alloc((size_t)2 * DI * DM * 2);
  __bf16* Winbf_b  = (__bf16*)alloc((size_t)2 * DI * DM * 2);
  __bf16* Wxbf_f   = (__bf16*)alloc((size_t)XDW * DI * 2);   // rows 80..127 zero
  __bf16* Wxbf_b   = (__bf16*)alloc((size_t)XDW * DI * 2);
  __bf16* Wdtbf_f  = (__bf16*)alloc((size_t)DI * 64 * 2);
  __bf16* Wdtbf_b  = (__bf16*)alloc((size_t)DI * 64 * 2);
  __bf16* Woutbf_f = (__bf16*)alloc((size_t)DM * DI * 2);
  __bf16* Woutbf_b = (__bf16*)alloc((size_t)DM * DI * 2);
  __bf16* Wprojbf  = (__bf16*)alloc((size_t)DM * (2 * DM) * 2);
  float*  xz_f     = (float*)alloc((size_t)MM * 2 * DI * 4);
  float*  xz_b     = (float*)alloc((size_t)MM * 2 * DI * 4);
  float*  xcf_f    = (float*)alloc((size_t)MM * DI * 4);
  float*  xcf_b    = (float*)alloc((size_t)MM * DI * 4);
  __bf16* xcb_f    = (__bf16*)alloc((size_t)MM * DI * 2);
  __bf16* xcb_b    = (__bf16*)alloc((size_t)MM * DI * 2);
  float*  xdbl_f   = (float*)alloc((size_t)MM * XDW * 4);    // cols 80..127 pad
  float*  xdbl_b   = (float*)alloc((size_t)MM * XDW * 4);
  __bf16* dtrbf_f  = (__bf16*)alloc((size_t)MM * 64 * 2);
  __bf16* dtrbf_b  = (__bf16*)alloc((size_t)MM * 64 * 2);
  float*  dtl_f    = (float*)alloc((size_t)MM * DI * 4);
  float*  dtl_b    = (float*)alloc((size_t)MM * DI * 4);
  float*  ys_f     = (float*)alloc((size_t)MM * DI * 4);
  float*  ys_b     = (float*)alloc((size_t)MM * DI * 4);
  __bf16* ybf_f    = (__bf16*)alloc((size_t)MM * DI * 2);
  __bf16* ybf_b    = (__bf16*)alloc((size_t)MM * DI * 2);
  __bf16* catbf    = (__bf16*)alloc((size_t)MM * DI * 2);

  dim3 blk(256);
  auto cvt = [&](const float* s, __bf16* d, long long n) {
    f32_to_bf16_k<<<dim3((unsigned)((n + 255) / 256)), blk, 0, stream>>>(s, d, n);
  };
  auto padc = [&](const float* s, __bf16* d, int rows, int sld, int srows,
                  int scols, int dcols) {
    long long n = (long long)rows * dcols;
    pad_cvt_k<<<dim3((unsigned)((n + 255) / 256)), blk, 0, stream>>>(s, d, rows, sld,
                                                                     srows, scols, dcols);
  };
  auto gemm = [&](const __bf16* A, const __bf16* Bw, void* C, const float* bias,
                  int M, int N, int K, int lda, int ldb, int ldc, int obf) {
    dim3 grid(N / 128, M / 128);
    gemm_bf16_wmma_k<<<grid, blk, 0, stream>>>(A, Bw, C, bias, M, N, K, lda, ldb, ldc, obf);
  };

  // 1) precision conversion of activations + weights (with padding)
  cvt(x, xbf, (long long)MM * DM);
  cvt(Win_f, Winbf_f, (long long)2 * DI * DM);
  cvt(Win_b, Winbf_b, (long long)2 * DI * DM);
  padc(Wx_f, Wxbf_f, XDW, DI, RK + 2 * DS, DI, DI);   // row-pad 80 -> 128
  padc(Wx_b, Wxbf_b, XDW, DI, RK + 2 * DS, DI, DI);
  cvt(Wout_f, Woutbf_f, (long long)DM * DI);
  cvt(Wout_b, Woutbf_b, (long long)DM * DI);
  cvt(Wproj, Wprojbf, (long long)DM * 2 * DM);
  padc(Wdt_f, Wdtbf_f, DI, RK, DI, RK, 64);           // col-pad 48 -> 64
  padc(Wdt_b, Wdtbf_b, DI, RK, DI, RK, 64);

  // 2) input projection GEMMs (shared non-reversed x; reversal folded later)
  gemm(xbf, Winbf_f, xz_f, nullptr, MM, 2 * DI, DM, DM, DM, 2 * DI, 0);
  gemm(xbf, Winbf_b, xz_b, nullptr, MM, 2 * DI, DM, DM, DM, 2 * DI, 0);

  // 3) causal depthwise conv + SiLU (both directions)
  {
    long long n = (long long)2 * BB * LL * DI;
    conv_silu_k<<<dim3((unsigned)((n + 255) / 256)), blk, 0, stream>>>(
        xz_f, xz_b, cw_f, cb_f, cw_b, cb_b, xcf_f, xcf_b, xcb_f, xcb_b);
  }

  // 4) x_dbl = xc @ W_x^T   (padded to N = 128; cols 80..127 unused)
  gemm(xcb_f, Wxbf_f, xdbl_f, nullptr, MM, XDW, DI, DI, DI, XDW, 0);
  gemm(xcb_b, Wxbf_b, xdbl_b, nullptr, MM, XDW, DI, DI, DI, XDW, 0);

  // 5) dt projection: pad dt_r K=48 -> 64, GEMM to (MM, DI)
  padc(xdbl_f, dtrbf_f, MM, XDW, MM, RK, 64);
  padc(xdbl_b, dtrbf_b, MM, XDW, MM, RK, 64);
  gemm(dtrbf_f, Wdtbf_f, dtl_f, nullptr, MM, DI, 64, 64, 64, DI, 0);
  gemm(dtrbf_b, Wdtbf_b, dtl_b, nullptr, MM, DI, 64, 64, 64, DI, 0);

  // 6) selective scan (sequential over L; both dirs + batches in grid)
  scan_k<<<dim3(DI / 256, BB, 2), blk, 0, stream>>>(
      xdbl_f, xdbl_b, dtl_f, dtl_b, xcf_f, xcf_b,
      Alog_f, bdt_f, Alog_b, bdt_b, ys_f, ys_b);

  // 7) gating + D skip + un-reversal, bf16 out
  {
    long long n = (long long)2 * BB * LL * DI;
    gate_k<<<dim3((unsigned)((n + 255) / 256)), blk, 0, stream>>>(
        ys_f, ys_b, xcf_f, xcf_b, xz_f, xz_b, D_f, D_b, ybf_f, ybf_b);
  }

  // 8) output projections write bf16 directly into concat halves
  gemm(ybf_f, Woutbf_f, catbf,      nullptr, MM, DM, DI, DI, DI, 2 * DM, 1);
  gemm(ybf_b, Woutbf_b, catbf + DM, nullptr, MM, DM, DI, DI, DI, 2 * DM, 1);

  // 9) final projection + bias -> fp32 output (2,2048,768)
  gemm(catbf, Wprojbf, (float*)d_out, bproj, MM, DM, 2 * DM, 2 * DM, 2 * DM, DM, 0);
}
